// GPT_89575837925710
// MI455X (gfx1250) — matrix operand order repaired and checked by reading, not compiled, not run on hardware
//
#include <hip/hip_runtime.h>
#include <hip/hip_bf16.h>
#include <math.h>

typedef __attribute__((ext_vector_type(16))) _Float16 v16h;
typedef __attribute__((ext_vector_type(4)))  _Float16 v4h;
typedef __attribute__((ext_vector_type(8)))  float    v8f;
typedef __attribute__((ext_vector_type(4)))  float    f32x4;

#define V_    50257
#define C_    768
#define T_    1024
#define B_    2
#define NH_   12
#define HD_   64
#define NL_   12
#define MROWS (B_ * T_)   // 2048

#define BM 128
#define BN 128
#define BK 32
#define LPAD 16           // pad LDS rows to 48 halves = 96B (16B-aligned stride)

// ---------------------------------------------------------------------------
// Embedding: x[b,t,:] = wte[idx[b,t],:] + wpe[t,:]
// ---------------------------------------------------------------------------
__global__ __launch_bounds__(256)
void embed_kernel(const int* __restrict__ idx, const float* __restrict__ wte,
                  const float* __restrict__ wpe, float* __restrict__ x) {
    int row = blockIdx.x;              // 0..2047  (b*T + t)
    int t   = row % T_;
    int tok = idx[row];
    const float* wt = wte + (size_t)tok * C_;
    const float* wp = wpe + (size_t)t * C_;
    float* xo = x + (size_t)row * C_;
    for (int c = threadIdx.x; c < C_; c += 256) xo[c] = wt[c] + wp[c];
}

// ---------------------------------------------------------------------------
// One-shot f32 -> f16 weight conversion (vectorized, grid-strided)
// ---------------------------------------------------------------------------
__global__ __launch_bounds__(256)
void cvt_f16_kernel(const float* __restrict__ src, _Float16* __restrict__ dst,
                    long n4) {
    long i      = (long)blockIdx.x * 256 + threadIdx.x;
    long stride = (long)gridDim.x * 256;
    for (; i < n4; i += stride) {
        f32x4 v = ((const f32x4*)src)[i];
        v4h h;
        h[0] = (_Float16)v[0]; h[1] = (_Float16)v[1];
        h[2] = (_Float16)v[2]; h[3] = (_Float16)v[3];
        ((v4h*)dst)[i] = h;
    }
}

// ---------------------------------------------------------------------------
// LayerNorm over last dim (C=768), one block per row
// ---------------------------------------------------------------------------
__global__ __launch_bounds__(256)
void layernorm_kernel(const float* __restrict__ x, const float* __restrict__ g,
                      const float* __restrict__ b, float* __restrict__ out) {
    __shared__ float red[256];
    int row = blockIdx.x;
    const float* xr = x + (size_t)row * C_;
    float* outr = out + (size_t)row * C_;

    float s = 0.f;
    for (int c = threadIdx.x; c < C_; c += 256) s += xr[c];
    red[threadIdx.x] = s; __syncthreads();
    for (int o = 128; o > 0; o >>= 1) {
        if (threadIdx.x < o) red[threadIdx.x] += red[threadIdx.x + o];
        __syncthreads();
    }
    float mean = red[0] * (1.0f / C_);
    __syncthreads();

    float v = 0.f;
    for (int c = threadIdx.x; c < C_; c += 256) { float d = xr[c] - mean; v += d * d; }
    red[threadIdx.x] = v; __syncthreads();
    for (int o = 128; o > 0; o >>= 1) {
        if (threadIdx.x < o) red[threadIdx.x] += red[threadIdx.x + o];
        __syncthreads();
    }
    float inv = rsqrtf(red[0] * (1.0f / C_) + 1e-5f);

    for (int c = threadIdx.x; c < C_; c += 256)
        outr[c] = (xr[c] - mean) * inv * g[c] + b[c];
}

// ---------------------------------------------------------------------------
// WMMA GEMM: out[M,N] = A[M,K] @ W16[K,N] (+bias) (gelu?) (+res)
// - A (f32 activations) staged f16 via v_cvt_pk (float4 load -> ds_store_b64)
// - W is PRE-CONVERTED f16: staging is 4 coalesced d16 loads (4 K-rows of one
//   column) packed straight into half-regs -> one transposed ds_store_b64.
//   No cvt VALU, half the weight traffic.
// - B fragments contiguous in LDS (Bt[col][k]) -> two ds_load_b128, same as A.
// - M always a multiple of BM (2048): no M guards.
// - NGUARD only for lm-head edge tile; branchless clamp+select, no execz.
// - HAS_BIAS/HAS_RES compile-time: straight-line epilogue.
// Block tile 128x128x32, 8 waves, each wave = 64x32 patch (4x2 WMMA tiles).
// ---------------------------------------------------------------------------
template <bool GELU, bool NGUARD, bool HAS_BIAS, bool HAS_RES>
__global__ __launch_bounds__(256)
void gemm_kernel(const float* __restrict__ A, const _Float16* __restrict__ W,
                 const float* __restrict__ bias, const float* __restrict__ res,
                 float* __restrict__ out, int M, int N, int K) {
    __shared__ _Float16 As[BM][BK + LPAD];   // 12 KB, 96B row stride
    __shared__ _Float16 Bt[BN][BK + LPAD];   // 12 KB, transposed tile

    const int tid  = threadIdx.x;
    const int lane = tid & 31;
    const int wave = tid >> 5;
    const int wm   = wave >> 2;    // 0..1 : 64-row band
    const int wn   = wave & 3;     // 0..3 : 32-col band
    const int bm   = blockIdx.y * BM;
    const int bn   = blockIdx.x * BN;

    v8f acc[4][2] = {};

    for (int k0 = 0; k0 < K; k0 += BK) {
        // ---- stage A tile: 128x32 f32, one float4 per thread x4 chunks ----
        #pragma unroll
        for (int c = 0; c < 4; ++c) {
            int i  = c * 256 + tid;          // 1024 chunks of 4 elements
            int ar = i >> 3;                 // 8 chunks per 32-wide row
            int ac = (i & 7) * 4;
            f32x4 av = *(const f32x4*)(A + (size_t)(bm + ar) * K + (k0 + ac));
            v4h hv;
            hv[0] = (_Float16)av[0]; hv[1] = (_Float16)av[1];
            hv[2] = (_Float16)av[2]; hv[3] = (_Float16)av[3];
            *(v4h*)&As[ar][ac] = hv;
        }
        // ---- stage B tile transposed: 4 K-rows of one column, f16 direct ----
        #pragma unroll
        for (int c = 0; c < 4; ++c) {
            int i   = c * 256 + tid;         // 1024 (col, kgroup) pairs
            int col = i & 127;
            int kr  = (i >> 7) * 4;
            int gc  = bn + col;
            v4h hv;
            if (NGUARD) {
                int gcs = gc < N ? gc : (N - 1);     // branchless clamp
                const _Float16* wp = W + (size_t)(k0 + kr) * N + gcs;
                _Float16 z = (_Float16)0.f;
                hv[0] = gc < N ? wp[0]             : z;
                hv[1] = gc < N ? wp[(size_t)N]     : z;
                hv[2] = gc < N ? wp[(size_t)2 * N] : z;
                hv[3] = gc < N ? wp[(size_t)3 * N] : z;
            } else {
                const _Float16* wp = W + (size_t)(k0 + kr) * N + gc;
                hv[0] = wp[0];             hv[1] = wp[(size_t)N];
                hv[2] = wp[(size_t)2 * N]; hv[3] = wp[(size_t)3 * N];
            }
            *(v4h*)&Bt[col][kr] = hv;
        }
        __syncthreads();

        // ---- B fragments: two 16B runs per lane (K=kb+0..7, 16+kb+0..7) -> 2x ds_load_b128 ----
        // 16-bit B 32x16 layout: lane n=l&15; kb=(l>>4)*8; elems 0..7 -> K=kb+e ; elems 8..15 -> K=16+kb+(e-8)
        v16h bfrag[2];
        {
            int n  = lane & 15;
            int kb = (lane >> 4) * 8;
            #pragma unroll
            for (int nj = 0; nj < 2; ++nj) {
                int ncol = wn * 32 + nj * 16 + n;
                #pragma unroll
                for (int e = 0; e < 16; ++e) bfrag[nj][e] = Bt[ncol][(e < 8 ? 0 : 8) + kb + e];
            }
        }
        // ---- A fragments + 8 WMMAs ----
        // 16-bit A 16x32 layout: lane m=l&15; kb=(l>>4)*8;
        // elems 0..7 -> K=kb+e ; elems 8..15 -> K=16+kb+(e-8)
        #pragma unroll
        for (int mi = 0; mi < 4; ++mi) {
            v16h afrag;
            int mrow = wm * 64 + mi * 16 + (lane & 15);
            int kb   = (lane >> 4) * 8;
            #pragma unroll
            for (int e = 0; e < 8; ++e) {
                afrag[e]     = As[mrow][kb + e];
                afrag[8 + e] = As[mrow][16 + kb + e];
            }
            #pragma unroll
            for (int nj = 0; nj < 2; ++nj) {
                acc[mi][nj] = __builtin_amdgcn_wmma_f32_16x16x32_f16(
                    false, afrag, false, bfrag[nj],
                    (short)0, acc[mi][nj], false, false);
            }
        }
        __syncthreads();
    }

    // ---- epilogue: C/D 16x16 f32 layout: lane n=l&15, VGPR v -> m=(l>>4)*8+v
    int rbase = bm + wm * 64 + (lane >> 4) * 8;
    int cbase = bn + wn * 32 + (lane & 15);
    #pragma unroll
    for (int nj = 0; nj < 2; ++nj) {
        int col = cbase + nj * 16;
        if (NGUARD && col >= N) continue;
        float bv = HAS_BIAS ? bias[col] : 0.f;   // hoisted: invariant per column
        #pragma unroll
        for (int mi = 0; mi < 4; ++mi) {
            #pragma unroll
            for (int v = 0; v < 8; ++v) {
                int row = rbase + mi * 16 + v;
                float val = acc[mi][nj][v] + bv;
                if (GELU)    val = 0.5f * val * (1.0f + erff(val * 0.70710678f));
                if (HAS_RES) val += res[(size_t)row * N + col];
                out[(size_t)row * N + col] = val;
            }
        }
    }
}

// ---------------------------------------------------------------------------
// Causal attention, one wave32 per (b,h,q). Online softmax per lane over its
// strided key set, then shfl_xor butterfly merge of (m, s, acc[64]).
// qkv row layout per head h: [q(64) | k(64) | v(64)] at offset h*192.
// All q/k/v rows are 16B-aligned -> explicit f32x4 (global_load_b128).
// ---------------------------------------------------------------------------
__global__ __launch_bounds__(32)
void attention_kernel(const float* __restrict__ qkv, float* __restrict__ y) {
    int q    = blockIdx.x & (T_ - 1);
    int h    = (blockIdx.x >> 10) % NH_;
    int b    = blockIdx.x / (T_ * NH_);
    int lane = threadIdx.x;

    const float* base = qkv + (size_t)(b * T_) * (3 * C_) + h * (3 * HD_);
    const f32x4* q4   = (const f32x4*)(base + (size_t)q * (3 * C_));

    f32x4 qv[HD_ / 4];
    #pragma unroll
    for (int d = 0; d < HD_ / 4; ++d) {
        f32x4 t = q4[d];
        qv[d][0] = t[0] * 0.125f; qv[d][1] = t[1] * 0.125f;   // 1/sqrt(64)
        qv[d][2] = t[2] * 0.125f; qv[d][3] = t[3] * 0.125f;
    }

    float m = -1e30f, s = 0.f;
    f32x4 acc[HD_ / 4];
    #pragma unroll
    for (int d = 0; d < HD_ / 4; ++d) acc[d] = (f32x4){0.f, 0.f, 0.f, 0.f};

    for (int j = lane; j <= q; j += 32) {
        const f32x4* k4 = (const f32x4*)(base + (size_t)j * (3 * C_) + HD_);
        float dot = 0.f;
        #pragma unroll
        for (int d = 0; d < HD_ / 4; ++d) {
            f32x4 kk = k4[d];
            dot += qv[d][0] * kk[0] + qv[d][1] * kk[1]
                 + qv[d][2] * kk[2] + qv[d][3] * kk[3];
        }
        float mn = fmaxf(m, dot);
        float sc = __expf(m - mn);
        float p  = __expf(dot - mn);
        s = s * sc + p;
        const f32x4* v4 = k4 + HD_ / 4;
        #pragma unroll
        for (int d = 0; d < HD_ / 4; ++d) {
            f32x4 vv = v4[d];
            acc[d][0] = acc[d][0] * sc + p * vv[0];
            acc[d][1] = acc[d][1] * sc + p * vv[1];
            acc[d][2] = acc[d][2] * sc + p * vv[2];
            acc[d][3] = acc[d][3] * sc + p * vv[3];
        }
        m = mn;
    }

    // butterfly merge across the wave (all lanes converge to full result)
    #pragma unroll
    for (int off = 16; off > 0; off >>= 1) {
        float mo = __shfl_xor(m, off, 32);
        float so = __shfl_xor(s, off, 32);
        float mn = fmaxf(m, mo);
        float sa = __expf(m - mn);
        float sb = __expf(mo - mn);
        s = s * sa + so * sb;
        #pragma unroll
        for (int d = 0; d < HD_ / 4; ++d) {
            #pragma unroll
            for (int e = 0; e < 4; ++e) {
                float ao = __shfl_xor(acc[d][e], off, 32);
                acc[d][e] = acc[d][e] * sa + ao * sb;
            }
        }
        m = mn;
    }

    float inv = 1.f / s;
    f32x4* yrow = (f32x4*)(y + (size_t)(b * T_ + q) * C_ + h * HD_);
    if (lane == 0) {
        #pragma unroll
        for (int d = 0; d < HD_ / 4; ++d) {
            f32x4 o;
            o[0] = acc[d][0] * inv; o[1] = acc[d][1] * inv;
            o[2] = acc[d][2] * inv; o[3] = acc[d][3] * inv;
            yrow[d] = o;
        }
    }
}

// ---------------------------------------------------------------------------
// Per-row NLL over V=50257 logits (one block per row), then scalar reduce
// ---------------------------------------------------------------------------
__global__ __launch_bounds__(256)
void nll_kernel(const float* __restrict__ logits, const int* __restrict__ tgt,
                float* __restrict__ nll) {
    __shared__ float red[256];
    int row = blockIdx.x;
    const float* lr = logits + (size_t)row * V_;

    float mx = -1e30f;
    for (int c = threadIdx.x; c < V_; c += 256) mx = fmaxf(mx, lr[c]);
    red[threadIdx.x] = mx; __syncthreads();
    for (int o = 128; o > 0; o >>= 1) {
        if (threadIdx.x < o) red[threadIdx.x] = fmaxf(red[threadIdx.x], red[threadIdx.x + o]);
        __syncthreads();
    }
    mx = red[0]; __syncthreads();

    float s = 0.f;
    for (int c = threadIdx.x; c < V_; c += 256) s += __expf(lr[c] - mx);
    red[threadIdx.x] = s; __syncthreads();
    for (int o = 128; o > 0; o >>= 1) {
        if (threadIdx.x < o) red[threadIdx.x] += red[threadIdx.x + o];
        __syncthreads();
    }
    if (threadIdx.x == 0) {
        int t = tgt[row];
        nll[row] = (t != -1) ? ((mx + __logf(red[0])) - lr[t]) : 0.f;
    }
}

__global__ __launch_bounds__(256)
void loss_kernel(const float* __restrict__ nll, const int* __restrict__ tgt,
                 float* __restrict__ loss_out) {
    __shared__ float red[256];
    __shared__ int   cnt[256];
    float s = 0.f; int c = 0;
    for (int i = threadIdx.x; i < MROWS; i += 256) { s += nll[i]; c += (tgt[i] != -1); }
    red[threadIdx.x] = s; cnt[threadIdx.x] = c; __syncthreads();
    for (int o = 128; o > 0; o >>= 1) {
        if (threadIdx.x < o) {
            red[threadIdx.x] += red[threadIdx.x + o];
            cnt[threadIdx.x] += cnt[threadIdx.x + o];
        }
        __syncthreads();
    }
    if (threadIdx.x == 0) {
        int n = cnt[0] > 0 ? cnt[0] : 1;
        loss_out[0] = red[0] / (float)n;
    }
}

// ---------------------------------------------------------------------------
// Orchestration. Input order: idx, targets, wte, wpe, 12 x {ln1_g, ln1_b,
// w_in, b_in, w_out, b_out, ln2_g, ln2_b, w_fc, b_fc, w_proj, b_proj},
// ln_f_g, ln_f_b, w_lm.  d_out = [logits (2048*50257 f32) | loss (1 f32)].
// Weights are converted f32->f16 into a reusable ws buffer per layer (stream
// ordering makes the reuse safe); lm-head weight converted after the loop.
// ---------------------------------------------------------------------------
extern "C" void kernel_launch(void* const* d_in, const int* in_sizes, int n_in,
                              void* d_out, int out_size, void* d_ws, size_t ws_size,
                              hipStream_t stream) {
    (void)in_sizes; (void)n_in; (void)out_size; (void)ws_size;

    const int*   idx     = (const int*)d_in[0];
    const int*   targets = (const int*)d_in[1];
    const float* wte     = (const float*)d_in[2];
    const float* wpe     = (const float*)d_in[3];
    const float* ln_f_g  = (const float*)d_in[4 + 12 * NL_];
    const float* ln_f_b  = (const float*)d_in[5 + 12 * NL_];
    const float* w_lm    = (const float*)d_in[6 + 12 * NL_];

    float* ws = (float*)d_ws;
    size_t o = 0;
    float* x   = ws + o;  o += (size_t)MROWS * C_;        // 6.3 MB
    float* h   = ws + o;  o += (size_t)MROWS * C_;        // 6.3 MB
    float* qkv = ws + o;  o += (size_t)MROWS * 3 * C_;    // 18.9 MB
    float* yb  = ws + o;  o += (size_t)MROWS * C_;        // 6.3 MB
    float* mb  = ws + o;  o += (size_t)MROWS * 4 * C_;    // 25.2 MB
    float* nll = ws + o;  o += (size_t)MROWS;             // 8 KB

    // f16 weight cache (reused each layer; sized for the largest user: w_lm)
    const size_t W_IN_SZ   = (size_t)C_ * 3 * C_;   // 1,769,472
    const size_t W_OUT_SZ  = (size_t)C_ * C_;       //   589,824
    const size_t W_FC_SZ   = (size_t)C_ * 4 * C_;   // 2,359,296
    _Float16* w16      = (_Float16*)(ws + o);       // 77 MB region
    _Float16* w_in16   = w16;
    _Float16* w_out16  = w_in16  + W_IN_SZ;
    _Float16* w_fc16   = w_out16 + W_OUT_SZ;
    _Float16* w_proj16 = w_fc16  + W_FC_SZ;

    float* logits = (float*)d_out;
    float* loss   = logits + (size_t)MROWS * V_;

    dim3 blk(256);
    const dim3 gCVT(1024);
    const dim3 gQKV(3 * C_ / BN, MROWS / BM);          // 18 x 16
    const dim3 gC  (C_ / BN, MROWS / BM);              //  6 x 16
    const dim3 gFC (4 * C_ / BN, MROWS / BM);          // 24 x 16
    const dim3 gLM ((V_ + BN - 1) / BN, MROWS / BM);   // 393 x 16

    embed_kernel<<<MROWS, blk, 0, stream>>>(idx, wte, wpe, x);

    for (int l = 0; l < NL_; ++l) {
        const float* const* P = (const float* const*)(d_in + 4 + l * 12);
        // weights -> f16 cache (stream-ordered, safe to reuse buffer per layer)
        cvt_f16_kernel<<<gCVT, blk, 0, stream>>>(P[2],  w_in16,   (long)(W_IN_SZ  / 4));
        cvt_f16_kernel<<<gCVT, blk, 0, stream>>>(P[4],  w_out16,  (long)(W_OUT_SZ / 4));
        cvt_f16_kernel<<<gCVT, blk, 0, stream>>>(P[8],  w_fc16,   (long)(W_FC_SZ  / 4));
        cvt_f16_kernel<<<gCVT, blk, 0, stream>>>(P[10], w_proj16, (long)(W_FC_SZ  / 4));
        // h = LN1(x)
        layernorm_kernel<<<MROWS, blk, 0, stream>>>(x, P[0], P[1], h);
        // qkv = h @ w_in + b_in
        gemm_kernel<false, false, true, false><<<gQKV, blk, 0, stream>>>(
            h, w_in16, P[3], nullptr, qkv, MROWS, 3 * C_, C_);
        // y = causal softmax attention
        attention_kernel<<<B_ * NH_ * T_, 32, 0, stream>>>(qkv, yb);
        // x = x + y @ w_out + b_out   (in-place residual)
        gemm_kernel<false, false, true, true><<<gC, blk, 0, stream>>>(
            yb, w_out16, P[5], x, x, MROWS, C_, C_);
        // h = LN2(x)
        layernorm_kernel<<<MROWS, blk, 0, stream>>>(x, P[6], P[7], h);
        // m = gelu(h @ w_fc + b_fc)
        gemm_kernel<true, false, true, false><<<gFC, blk, 0, stream>>>(
            h, w_fc16, P[9], nullptr, mb, MROWS, 4 * C_, C_);
        // x = h + m @ w_proj + b_proj   (residual is h, per reference!)
        gemm_kernel<false, false, true, true><<<gC, blk, 0, stream>>>(
            mb, w_proj16, P[11], h, x, MROWS, C_, 4 * C_);
    }

    // h = LN_f(x);  logits = h @ w_lm   (N=50257 needs edge guards)
    layernorm_kernel<<<MROWS, blk, 0, stream>>>(x, ln_f_g, ln_f_b, h);
    cvt_f16_kernel<<<gCVT, blk, 0, stream>>>(w_lm, w16, (long)((size_t)C_ * V_ / 4));
    gemm_kernel<false, true, false, false><<<gLM, blk, 0, stream>>>(
        h, w16, nullptr, nullptr, logits, MROWS, V_, C_);

    // cross-entropy
    nll_kernel<<<MROWS, blk, 0, stream>>>(logits, targets, nll);
    loss_kernel<<<1, blk, 0, stream>>>(nll, targets, loss);
}
